// Attention_63574105915469
// MI455X (gfx1250) — matrix operand order repaired and checked by reading, not compiled, not run on hardware
//
#include <hip/hip_runtime.h>
#include <hip/hip_bf16.h>
#include <cstdint>

// ---- CDNA5 WMMA types ------------------------------------------------------
typedef __attribute__((ext_vector_type(16))) __bf16 v16bf;
typedef __attribute__((ext_vector_type(8)))  float  v8f;

#define DEV_INLINE __device__ __forceinline__

// Problem constants (from reference)
#define BB 4
#define NN 1024
#define DQ 256
#define HH 8
#define DD 32
#define MROWS (BB * NN)      // 4096
#define QSCALE 0.17677669529663689f  // 32^-0.5

DEV_INLINE v8f wmma_bf16(v16bf a, v16bf b, v8f c) {
    return __builtin_amdgcn_wmma_f32_16x16x32_bf16(false, a, false, b, (short)0, c,
                                                   false, false);
}

// ============================================================================
// Kernel 1: projections  Y[M,256] = X[M,256] @ W[256,256]^T  (bf16 WMMA)
//   mode 0: Q  -> Qh bf16 [B,H,N,D], scaled by QSCALE
//   mode 1: K  -> Kh bf16 [B,H,N,D]
//   mode 2: V  -> Vt bf16 [B,H,D,N]   (transposed for PV B-operand)
//   mode 3: G  -> gates f32 [M,256] = sigmoid(y + bg)
// one wave per 16x16 output tile; 4 waves / block; 4096 tiles
// ============================================================================
__global__ void __launch_bounds__(128)
proj_kernel(const float* __restrict__ X, const float* __restrict__ W,
            const float* __restrict__ bvec, __hip_bfloat16* __restrict__ outB,
            float* __restrict__ outF, int mode) {
    const int lane = threadIdx.x & 31;
    const int hl = lane >> 4;        // lane half
    const int lr = lane & 15;
    const int wid = blockIdx.x * 4 + (threadIdx.x >> 5);
    const int tm = wid >> 4;         // 0..255  (M tile)
    const int tn = wid & 15;         // 0..15   (feature tile)

    const int arow = tm * 16 + lr;   // A-matrix row (striped over lanes)
    const int bcol = tn * 16 + lr;   // B-matrix col = out feature

    v8f acc = {};
#pragma unroll
    for (int k0 = 0; k0 < DQ; k0 += 32) {
        const float* ap = X + (size_t)arow * DQ + k0 + hl * 16;
        v16bf a;
#pragma unroll
        for (int e = 0; e < 8; ++e) {
            a[e]     = (__bf16)ap[e];
            a[e + 8] = (__bf16)ap[e + 8];
        }
        const float* bp = W + (size_t)bcol * DQ + k0 + hl * 16;
        v16bf b;
#pragma unroll
        for (int e = 0; e < 16; ++e) b[e] = (__bf16)bp[e];
        acc = wmma_bf16(a, b, acc);
    }

    // C layout: lane holds column feature bcol; VGPR r holds row tm*16 + r + hl*8
#pragma unroll
    for (int r = 0; r < 8; ++r) {
        const int grow = tm * 16 + r + hl * 8;      // global row in [0,4096)
        const int b    = grow >> 10;
        const int n    = grow & (NN - 1);
        const int h    = bcol >> 5;
        const int d    = bcol & (DD - 1);
        float v = acc[r];
        if (mode == 0) {
            outB[(((size_t)b * HH + h) * NN + n) * DD + d] =
                __float2bfloat16(v * QSCALE);
        } else if (mode == 1) {
            outB[(((size_t)b * HH + h) * NN + n) * DD + d] = __float2bfloat16(v);
        } else if (mode == 2) {
            outB[(((size_t)b * HH + h) * DD + d) * NN + n] = __float2bfloat16(v);
        } else {
            float g = v + bvec[bcol];
            outF[(size_t)grow * DQ + bcol] = 1.0f / (1.0f + __expf(-g));
        }
    }
}

// ============================================================================
// Kernel 2: attention with online softmax (flash style), one wave per
// 16-row i-tile of one (b,h). 2048 waves, 8 waves/block.
// ============================================================================
__global__ void __launch_bounds__(256)
attn_kernel(const __hip_bfloat16* __restrict__ Qh,
            const __hip_bfloat16* __restrict__ Kh,
            const __hip_bfloat16* __restrict__ Vt,
            const float* __restrict__ bias,   // [B,H,N,N]
            const float* __restrict__ mask,   // [B,N]
            const float* __restrict__ gates,  // [M,256] (sigmoid applied)
            __hip_bfloat16* __restrict__ Og)  // [M,256] bf16
{
    __shared__ __bf16 pbuf[8][16 * 32];     // per-wave P tile (C->A transpose)

    const int lane = threadIdx.x & 31;
    const int hl = lane >> 4;
    const int lr = lane & 15;
    const int wslot = threadIdx.x >> 5;
    const int wid = blockIdx.x * 8 + wslot;
    const int bh = wid >> 6;                 // 0..31  (b*H+h)
    const int it = wid & 63;                 // i tile within (b,h)
    const int b = bh >> 3, h = bh & 7;
    const int i0 = it * 16;

    const __bf16* QhB = (const __bf16*)Qh;
    const __bf16* KhB = (const __bf16*)Kh;
    const __bf16* VtB = (const __bf16*)Vt;

    // Q A-tile (reused for all j): lane row i0+lr, k over D=32
    v16bf qa;
    {
        const __bf16* qp = QhB + (((size_t)bh * NN) + i0 + lr) * DD + hl * 16;
#pragma unroll
        for (int e = 0; e < 8; ++e) { qa[e] = qp[e]; qa[e + 8] = qp[e + 8]; }
    }

    float m_run[8], l_run[8];
    v8f O0 = {}, O1 = {};
#pragma unroll
    for (int r = 0; r < 8; ++r) { m_run[r] = -3.0e38f; l_run[r] = 0.0f; }

    __bf16* pb = pbuf[wslot];

    for (int j0 = 0; j0 < NN; j0 += 32) {
        // ---- S = Q K^T : two 16x16 tiles (cols j0..j0+15, j0+16..j0+31) ----
        v16bf bk0, bk1;
        {
            const __bf16* kp0 = KhB + (((size_t)bh * NN) + j0 + lr) * DD + hl * 16;
            const __bf16* kp1 = kp0 + 16 * DD;
#pragma unroll
            for (int e = 0; e < 16; ++e) { bk0[e] = kp0[e]; bk1[e] = kp1[e]; }
        }
        v8f S0 = {}, S1 = {};
        S0 = wmma_bf16(qa, bk0, S0);
        S1 = wmma_bf16(qa, bk1, S1);

        // ---- bias (coalesced in j) + mask ----
        const float* bp = bias + (((size_t)bh * NN) + i0) * NN + j0;
        const float mj0 = (mask[b * NN + j0 + lr]      > 0.f) ? 0.f : -3.0e38f;
        const float mj1 = (mask[b * NN + j0 + 16 + lr] > 0.f) ? 0.f : -3.0e38f;
#pragma unroll
        for (int r = 0; r < 8; ++r) {
            const float* row = bp + (size_t)(r + hl * 8) * NN;
            S0[r] += row[lr] + mj0;
            S1[r] += row[16 + lr] + mj1;
        }

        // ---- row max over 32 cols (butterfly across 16-lane half) ----
        float mt[8];
#pragma unroll
        for (int r = 0; r < 8; ++r) mt[r] = fmaxf(S0[r], S1[r]);
#pragma unroll
        for (int off = 1; off < 16; off <<= 1) {
#pragma unroll
            for (int r = 0; r < 8; ++r)
                mt[r] = fmaxf(mt[r], __shfl_xor(mt[r], off, 32));
        }

        // ---- online softmax update ----
        float rs[8];
#pragma unroll
        for (int r = 0; r < 8; ++r) {
            const float mn = fmaxf(m_run[r], mt[r]);
            const float sc = __expf(m_run[r] - mn);
            m_run[r] = mn;
            S0[r] = __expf(S0[r] - mn);
            S1[r] = __expf(S1[r] - mn);
            rs[r] = S0[r] + S1[r];
            l_run[r] = l_run[r] * sc + 0.0f;   // sum added after reduction
            O0[r] *= sc;
            O1[r] *= sc;
        }
#pragma unroll
        for (int off = 1; off < 16; off <<= 1) {
#pragma unroll
            for (int r = 0; r < 8; ++r) rs[r] += __shfl_xor(rs[r], off, 32);
        }
#pragma unroll
        for (int r = 0; r < 8; ++r) l_run[r] += rs[r];

        // ---- P: C layout -> A layout via per-wave LDS tile (16x32 bf16) ----
#pragma unroll
        for (int r = 0; r < 8; ++r) {
            pb[(r + hl * 8) * 32 + lr]      = (__bf16)S0[r];
            pb[(r + hl * 8) * 32 + 16 + lr] = (__bf16)S1[r];
        }
        v16bf pa;
        {
            const __bf16* pr = pb + lr * 32 + hl * 16;
#pragma unroll
            for (int e = 0; e < 8; ++e) { pa[e] = pr[e]; pa[e + 8] = pr[e + 8]; }
        }

        // ---- O += P @ V : V^T gives contiguous j per lane ----
        v16bf bv0, bv1;
        {
            const __bf16* vp0 = VtB + (((size_t)bh * DD) + lr) * NN + j0 + hl * 16;
            const __bf16* vp1 = VtB + (((size_t)bh * DD) + 16 + lr) * NN + j0 + hl * 16;
#pragma unroll
            for (int e = 0; e < 16; ++e) { bv0[e] = vp0[e]; bv1[e] = vp1[e]; }
        }
        O0 = wmma_bf16(pa, bv0, O0);
        O1 = wmma_bf16(pa, bv1, O1);
    }

    // ---- epilogue: normalize, gate, store bf16 [M, H*D] ----
#pragma unroll
    for (int r = 0; r < 8; ++r) {
        const float inv = 1.0f / l_run[r];
        const int grow = b * NN + i0 + r + hl * 8;         // row in [0,4096)
        const int e0 = h * DD + lr;
        const float g0 = gates[(size_t)grow * DQ + e0];
        const float g1 = gates[(size_t)grow * DQ + e0 + 16];
        Og[(size_t)grow * DQ + e0]      = __float2bfloat16(O0[r] * inv * g0);
        Og[(size_t)grow * DQ + e0 + 16] = __float2bfloat16(O1[r] * inv * g1);
    }
}

// ============================================================================
// Kernel 3: output projection  Y[M,256] = Og[M,256] @ Wo[256,256]^T + bo
// ============================================================================
__global__ void __launch_bounds__(128)
outproj_kernel(const __hip_bfloat16* __restrict__ Og,
               const float* __restrict__ Wo, const float* __restrict__ bo,
               float* __restrict__ Y) {
    const int lane = threadIdx.x & 31;
    const int hl = lane >> 4;
    const int lr = lane & 15;
    const int wid = blockIdx.x * 4 + (threadIdx.x >> 5);
    const int tm = wid >> 4;
    const int tn = wid & 15;

    const __bf16* OgB = (const __bf16*)Og;
    const int arow = tm * 16 + lr;
    const int bcol = tn * 16 + lr;

    v8f acc = {};
#pragma unroll
    for (int k0 = 0; k0 < DQ; k0 += 32) {
        const __bf16* ap = OgB + (size_t)arow * DQ + k0 + hl * 16;
        v16bf a;
#pragma unroll
        for (int e = 0; e < 8; ++e) { a[e] = ap[e]; a[e + 8] = ap[e + 8]; }
        const float* bp = Wo + (size_t)bcol * DQ + k0 + hl * 16;
        v16bf bmat;
#pragma unroll
        for (int e = 0; e < 16; ++e) bmat[e] = (__bf16)bp[e];
        acc = wmma_bf16(a, bmat, acc);
    }
    const float bc = bo[bcol];
#pragma unroll
    for (int r = 0; r < 8; ++r) {
        const int grow = tm * 16 + r + hl * 8;
        Y[(size_t)grow * DQ + bcol] = acc[r] + bc;
    }
}

// ============================================================================
// host launcher
// ============================================================================
extern "C" void kernel_launch(void* const* d_in, const int* in_sizes, int n_in,
                              void* d_out, int out_size, void* d_ws, size_t ws_size,
                              hipStream_t stream) {
    const float* x         = (const float*)d_in[0];
    const float* mask      = (const float*)d_in[1];
    const float* attn_bias = (const float*)d_in[2];
    const float* Wq        = (const float*)d_in[3];
    const float* Wkv       = (const float*)d_in[4];
    const float* Wo        = (const float*)d_in[5];
    const float* bo        = (const float*)d_in[6];
    const float* Wg        = (const float*)d_in[7];
    const float* bg        = (const float*)d_in[8];

    char* ws = (char*)d_ws;
    const size_t szBF = (size_t)MROWS * DQ * sizeof(__hip_bfloat16);  // 2 MB
    __hip_bfloat16* Qh    = (__hip_bfloat16*)(ws);
    __hip_bfloat16* Kh    = (__hip_bfloat16*)(ws + szBF);
    __hip_bfloat16* Vt    = (__hip_bfloat16*)(ws + 2 * szBF);
    float*          gates = (float*)(ws + 3 * szBF);
    __hip_bfloat16* Og    = (__hip_bfloat16*)(ws + 3 * szBF +
                                              (size_t)MROWS * DQ * sizeof(float));

    // projections
    proj_kernel<<<1024, 128, 0, stream>>>(x, Wq,            nullptr, Qh, nullptr, 0);
    proj_kernel<<<1024, 128, 0, stream>>>(x, Wkv,           nullptr, Kh, nullptr, 1);
    proj_kernel<<<1024, 128, 0, stream>>>(x, Wkv + 256*256, nullptr, Vt, nullptr, 2);
    proj_kernel<<<1024, 128, 0, stream>>>(x, Wg,            bg, nullptr, gates, 3);
    // attention (flash-style)
    attn_kernel<<<256, 256, 0, stream>>>(Qh, Kh, Vt, attn_bias, mask, gates, Og);
    // output projection
    outproj_kernel<<<1024, 128, 0, stream>>>(Og, Wo, bo, (float*)d_out);
}
